// MuZeroNet_52355651338923
// MI455X (gfx1250) — matrix operand order repaired and checked by reading, not compiled, not run on hardware
//
#include <hip/hip_runtime.h>
#include <hip/hip_bf16.h>

// ---------------------------------------------------------------------------
// MuZero forward pass for gfx1250 (MI455X): bf16 WMMA GEMMs + routed top-1 MoE
// K-loop: GLOBAL_LOAD_ASYNC_TO_LDS_B128, triple-buffered LDS, 1 barrier/step.
// ---------------------------------------------------------------------------

typedef __attribute__((ext_vector_type(16))) __bf16 v16bf;
typedef __attribute__((ext_vector_type(8)))  float  v8f;

union FragB16 { v16bf v; uint4 q[2]; };

__device__ __forceinline__ unsigned short f2bf(float f) {
  unsigned int u = __builtin_bit_cast(unsigned int, f);
  u += 0x7fffu + ((u >> 16) & 1u);      // round-to-nearest-even
  return (unsigned short)(u >> 16);
}
__device__ __forceinline__ float bf2f(unsigned short h) {
  return __builtin_bit_cast(float, ((unsigned int)h) << 16);
}

// CDNA5 async memory->LDS (ASYNCcnt tracked, no VGPR staging).
__device__ __forceinline__ void async_ld_b128(unsigned ldsAddr,
                                              unsigned long long gaddr) {
  asm volatile("global_load_async_to_lds_b128 %0, %1, off"
               :: "v"(ldsAddr), "v"(gaddr) : "memory");
}
__device__ __forceinline__ void wait_async_le4() {
  asm volatile("s_wait_asynccnt 0x4" ::: "memory");
}
__device__ __forceinline__ void wait_async_le0() {
  asm volatile("s_wait_asynccnt 0x0" ::: "memory");
}

// ---------------------------------------------------------------------------
// Unified WMMA GEMM:  C[M,N] = act(A[M,K] @ Wt[N,K]^T + bias)
// MODE 0: plain rows. MODE 1: MoE, A rows gathered via rowIdx, compact out.
// MODE 2: MoE, A compact (segment) rows, compact out.
// Block: 256 thr = 8 waves; tile 128x128, BK=32; wave = 32x64 (2x4 WMMA frags)
// ---------------------------------------------------------------------------
template <int MODE, bool RELU>
__global__ __launch_bounds__(256)
void k_gemm(const unsigned short* __restrict__ A,
            const unsigned short* __restrict__ W,     // [ (E,) N, K ] bf16
            const float* __restrict__ bias,           // [ (E,) N ]
            float* __restrict__ outF,                 // [M,N] f32 or null
            unsigned short* __restrict__ outB,        // [M,N] bf16 or null
            int M, int N, int K,
            const int* __restrict__ rowIdx,           // MoE gather indices
            const int* __restrict__ route)            // [0..3]=cnt, [4..7]=off
{
  int off = 0, cnt = M;
  const unsigned short* Wp = W;
  const float* bp = bias;
  if (MODE != 0) {
    const int e = blockIdx.z;
    cnt = route[e];
    off = route[4 + e];
    Wp  = W + (size_t)e * N * K;
    bp  = bias + (size_t)e * N;
  }
  const int m0 = blockIdx.y * 128;
  if (m0 >= cnt) return;
  const int n0 = blockIdx.x * 128;

  // triple-buffered LDS tiles: 128 rows x (32+8 pad) halves each
  __shared__ __align__(16) unsigned short As[3][128 * 40];
  __shared__ __align__(16) unsigned short Bs[3][128 * 40];

  const int tid  = threadIdx.x;
  const int lane = tid & 31, wave = tid >> 5;
  const int wm = (wave & 3) * 32;   // wave M offset in tile
  const int wn = (wave >> 2) * 64;  // wave N offset in tile
  const int row = lane & 15, hi = lane >> 4;

  v8f acc[2][4] = {};

  // two load slots per thread: slot -> (row = slot>>2, 8-half chunk = slot&3)
  const int r0 = tid >> 2,          c0 = (tid & 3) * 8;
  const int r1 = (tid + 256) >> 2,  c1 = ((tid + 256) & 3) * 8;

  // resolve A source rows (fixed across K loop)
  int lmA0 = m0 + r0; if (lmA0 >= cnt) lmA0 = cnt - 1;
  int lmA1 = m0 + r1; if (lmA1 >= cnt) lmA1 = cnt - 1;
  const int gaA0 = (MODE == 1) ? rowIdx[off + lmA0] : (off + lmA0);
  const int gaA1 = (MODE == 1) ? rowIdx[off + lmA1] : (off + lmA1);
  int nr0 = n0 + r0; if (nr0 >= N) nr0 = N - 1;
  int nr1 = n0 + r1; if (nr1 >= N) nr1 = N - 1;

  // global byte addresses of this thread's four 16B chunks (k0 = 0)
  const unsigned long long gA0 =
      (unsigned long long)(uintptr_t)(A + (size_t)gaA0 * K + c0);
  const unsigned long long gA1 =
      (unsigned long long)(uintptr_t)(A + (size_t)gaA1 * K + c1);
  const unsigned long long gB0 =
      (unsigned long long)(uintptr_t)(Wp + (size_t)nr0 * K + c0);
  const unsigned long long gB1 =
      (unsigned long long)(uintptr_t)(Wp + (size_t)nr1 * K + c1);

  // LDS byte addresses of the destination slots (buffer 0)
  const unsigned lA0 = (unsigned)(uintptr_t)&As[0][r0 * 40 + c0];
  const unsigned lA1 = (unsigned)(uintptr_t)&As[0][r1 * 40 + c1];
  const unsigned lB0 = (unsigned)(uintptr_t)&Bs[0][r0 * 40 + c0];
  const unsigned lB1 = (unsigned)(uintptr_t)&Bs[0][r1 * 40 + c1];
  const unsigned bufStep = 128 * 40 * 2;   // bytes between buffers

  const int S = K / 32;   // k-steps (all K are multiples of 32)

  // one MMA step over LDS buffer bc
  auto mma_step = [&](int bc) {
    const unsigned short* Asb = &As[0][0] + bc * (128 * 40);
    const unsigned short* Bsb = &Bs[0][0] + bc * (128 * 40);
    FragB16 af[2], bfm[4];
#pragma unroll
    for (int mi = 0; mi < 2; ++mi) {
      const unsigned short* p = Asb + (wm + mi * 16 + row) * 40;
      af[mi].q[0] = *(const uint4*)(p + hi * 8);        // K = hi*8 .. +7
      af[mi].q[1] = *(const uint4*)(p + 16 + hi * 8);   // K = 16+hi*8 .. +7
    }
#pragma unroll
    for (int ni = 0; ni < 4; ++ni) {
      const unsigned short* p = Bsb + (wn + ni * 16 + row) * 40 + hi * 8;
      bfm[ni].q[0] = *(const uint4*)(p);                // K = hi*8 .. +7
      bfm[ni].q[1] = *(const uint4*)(p + 16);           // K = 16+hi*8 .. +7
    }
#pragma unroll
    for (int mi = 0; mi < 2; ++mi)
#pragma unroll
      for (int ni = 0; ni < 4; ++ni)
        acc[mi][ni] = __builtin_amdgcn_wmma_f32_16x16x32_bf16(
            false, af[mi].v, false, bfm[ni].v, (short)0, acc[mi][ni],
            false, false);
  };

  // prologue: tile 0 -> buffer 0
  async_ld_b128(lA0, gA0);
  async_ld_b128(lA1, gA1);
  async_ld_b128(lB0, gB0);
  async_ld_b128(lB1, gB1);

  // Steady state: issue tile s+1 into buf[(s+1)%3], wait for tile s (in-order
  // ASYNCcnt), one barrier, compute buf[s%3]. Safe with 3 buffers: issuing
  // tile s+1 overwrites the buffer last read at compute(s-2), and barrier(s-1)
  // (which follows compute(s-2) in every wave's program order) has passed.
  int bIss = 1;   // buffer receiving the next issued tile
  int bCmp = 0;   // buffer being computed
  for (int s = 0; s < S - 1; ++s) {
    const unsigned long long adv = (unsigned long long)(s + 1) * 64;  // bytes
    const unsigned lo = (unsigned)bIss * bufStep;
    async_ld_b128(lA0 + lo, gA0 + adv);
    async_ld_b128(lA1 + lo, gA1 + adv);
    async_ld_b128(lB0 + lo, gB0 + adv);
    async_ld_b128(lB1 + lo, gB1 + adv);
    wait_async_le4();    // tile s has landed; tile s+1 (4 ops) may remain
    __syncthreads();
    mma_step(bCmp);
    if (++bIss == 3) bIss = 0;
    if (++bCmp == 3) bCmp = 0;
  }
  wait_async_le0();      // last tile landed
  __syncthreads();
  mma_step(bCmp);

  // epilogue: c[j] holds (M = mi*16 + hi*8 + j, N = ni*16 + row)
#pragma unroll
  for (int ni = 0; ni < 4; ++ni) {
    const int gn = n0 + wn + ni * 16 + row;
    if (gn >= N) continue;
    const float bv = bp ? bp[gn] : 0.f;
#pragma unroll
    for (int mi = 0; mi < 2; ++mi) {
#pragma unroll
      for (int j = 0; j < 8; ++j) {
        const int gm = m0 + wm + mi * 16 + hi * 8 + j;
        if (gm >= cnt) continue;
        float v = acc[mi][ni][j] + bv;
        if (RELU) v = v > 0.f ? v : 0.f;
        const size_t o = (size_t)(off + gm) * N + gn;
        if (outF) outF[o] = v;
        if (outB) outB[o] = f2bf(v);
      }
    }
  }
}

// ---------------------------------------------------------------------------
// LayerNorm over D (mean/biased-var), optional relu, f32 and/or bf16 out
// ---------------------------------------------------------------------------
template <bool RELU>
__global__ __launch_bounds__(256)
void k_layernorm(const float* __restrict__ in, const float* __restrict__ g,
                 const float* __restrict__ b, int D,
                 float* __restrict__ outF, unsigned short* __restrict__ outB)
{
  const int rowi = blockIdx.x;
  const float* x = in + (size_t)rowi * D;
  const int tid = threadIdx.x, lane = tid & 31, wave = tid >> 5;
  float s = 0.f, s2 = 0.f;
  for (int i = tid; i < D; i += 256) { float v = x[i]; s += v; s2 += v * v; }
  for (int o = 16; o > 0; o >>= 1) {
    s += __shfl_xor(s, o, 32); s2 += __shfl_xor(s2, o, 32);
  }
  __shared__ float rs[8], rs2[8];
  if (lane == 0) { rs[wave] = s; rs2[wave] = s2; }
  __syncthreads();
  if (wave == 0) {
    s  = (lane < 8) ? rs[lane]  : 0.f;
    s2 = (lane < 8) ? rs2[lane] : 0.f;
    for (int o = 4; o > 0; o >>= 1) {
      s += __shfl_xor(s, o, 32); s2 += __shfl_xor(s2, o, 32);
    }
    if (lane == 0) { rs[0] = s; rs2[0] = s2; }
  }
  __syncthreads();
  const float mean = rs[0] / D;
  const float var  = rs2[0] / D - mean * mean;
  const float inv  = rsqrtf(var + 1e-5f);
  for (int i = tid; i < D; i += 256) {
    float v = (x[i] - mean) * inv * g[i] + b[i];
    if (RELU) v = fmaxf(v, 0.f);
    const size_t o = (size_t)rowi * D + i;
    if (outF) outF[o] = v;
    if (outB) outB[o] = f2bf(v);
  }
}

// x = relu(t + x) in place; optional second f32 copy; bf16 copy
__global__ void k_add_relu(const float* __restrict__ t, float* __restrict__ x,
                           float* __restrict__ outF2,
                           unsigned short* __restrict__ outB, int n)
{
  const int i = (blockIdx.x * 256 + threadIdx.x) * 4;
  if (i >= n) return;
  float4 tv = *(const float4*)(t + i);
  float4 xv = *(const float4*)(x + i);
  float4 r;
  r.x = fmaxf(tv.x + xv.x, 0.f); r.y = fmaxf(tv.y + xv.y, 0.f);
  r.z = fmaxf(tv.z + xv.z, 0.f); r.w = fmaxf(tv.w + xv.w, 0.f);
  *(float4*)(x + i) = r;
  if (outF2) *(float4*)(outF2 + i) = r;
  ushort4 h = make_ushort4(f2bf(r.x), f2bf(r.y), f2bf(r.z), f2bf(r.w));
  *(ushort4*)(outB + i) = h;
}

// x[row] = y_compact[p] + x[row]; bf16 copy to outB[row]
__global__ void k_scatter_add(const float* __restrict__ y, float* __restrict__ x,
                              unsigned short* __restrict__ outB,
                              const int* __restrict__ rowIdx, int D)
{
  const int p = blockIdx.x;
  const int r = rowIdx[p];
  const float* yp = y + (size_t)p * D;
  float* xp = x + (size_t)r * D;
  unsigned short* bp = outB + (size_t)r * D;
  for (int i = threadIdx.x * 4; i < D; i += 256 * 4) {
    float4 yv = *(const float4*)(yp + i);
    float4 xv = *(const float4*)(xp + i);
    float4 s;
    s.x = yv.x + xv.x; s.y = yv.y + xv.y; s.z = yv.z + xv.z; s.w = yv.w + xv.w;
    *(float4*)(xp + i) = s;
    ushort4 h = make_ushort4(f2bf(s.x), f2bf(s.y), f2bf(s.z), f2bf(s.w));
    *(ushort4*)(bp + i) = h;
  }
}

// gate: logits = x @ Wg[1024,4] + bg, top-1 -> expertId; counts via atomics
__global__ __launch_bounds__(256)
void k_gate(const float* __restrict__ x, const float* __restrict__ gw,
            const float* __restrict__ gb, int D,
            int* __restrict__ expertId, int* __restrict__ route)
{
  const int wave = threadIdx.x >> 5, lane = threadIdx.x & 31;
  const int row = blockIdx.x * 8 + wave;
  const float* xp = x + (size_t)row * D;
  float4 s = make_float4(0.f, 0.f, 0.f, 0.f);
  for (int k = lane; k < D; k += 32) {
    const float xv = xp[k];
    const float4 wv = *(const float4*)(gw + k * 4);
    s.x += xv * wv.x; s.y += xv * wv.y; s.z += xv * wv.z; s.w += xv * wv.w;
  }
  for (int o = 16; o > 0; o >>= 1) {
    s.x += __shfl_xor(s.x, o, 32); s.y += __shfl_xor(s.y, o, 32);
    s.z += __shfl_xor(s.z, o, 32); s.w += __shfl_xor(s.w, o, 32);
  }
  if (lane == 0) {
    const float l0 = s.x + gb[0], l1 = s.y + gb[1];
    const float l2 = s.z + gb[2], l3 = s.w + gb[3];
    int e = 0; float m = l0;
    if (l1 > m) { m = l1; e = 1; }
    if (l2 > m) { m = l2; e = 2; }
    if (l3 > m) { m = l3; e = 3; }
    expertId[row] = e;
    atomicAdd(&route[e], 1);
  }
}

__global__ void k_route_init(int* __restrict__ route) {
  if (threadIdx.x < 12) route[threadIdx.x] = 0;
}
__global__ void k_route_scan(int* __restrict__ route) {
  if (threadIdx.x == 0) {
    int c = 0;
    for (int e = 0; e < 4; ++e) { route[4 + e] = c; route[8 + e] = c; c += route[e]; }
  }
}
__global__ void k_route_scatter(const int* __restrict__ expertId,
                                int* __restrict__ route, int* __restrict__ rowIdx)
{
  const int row = blockIdx.x * 256 + threadIdx.x;
  const int e = expertId[row];
  const int pos = atomicAdd(&route[8 + e], 1);
  rowIdx[pos] = row;
}

// value head final: out[row] = dot(t_bf16[row,0:D], w) + b
__global__ __launch_bounds__(256)
void k_rowdot(const unsigned short* __restrict__ t, const float* __restrict__ w,
              const float* __restrict__ b, float* __restrict__ out, int D)
{
  const int wave = threadIdx.x >> 5, lane = threadIdx.x & 31;
  const int row = blockIdx.x * 8 + wave;
  const unsigned short* tp = t + (size_t)row * D;
  float s = 0.f;
  for (int k = lane; k < D; k += 32) s += bf2f(tp[k]) * w[k];
  for (int o = 16; o > 0; o >>= 1) s += __shfl_xor(s, o, 32);
  if (lane == 0) out[row] = s + b[0];
}

// weight prep: fp32 [K,N] -> bf16 transposed [N,K]
__global__ __launch_bounds__(256)
void k_transpose_w(const float* __restrict__ in, unsigned short* __restrict__ out,
                   int K, int N)
{
  __shared__ float tile[32][33];
  const int tx = threadIdx.x, ty = threadIdx.y;   // 32 x 8
  const int nb = blockIdx.x * 32, kb = blockIdx.y * 32;
#pragma unroll
  for (int i = 0; i < 4; ++i) {
    const int k = kb + ty + i * 8, n = nb + tx;
    tile[ty + i * 8][tx] = (k < K && n < N) ? in[(size_t)k * N + n] : 0.f;
  }
  __syncthreads();
#pragma unroll
  for (int i = 0; i < 4; ++i) {
    const int n = nb + ty + i * 8, k = kb + tx;
    if (n < N && k < K) out[(size_t)n * K + k] = f2bf(tile[tx][ty + i * 8]);
  }
}

__global__ void k_f32_to_bf16(const float* __restrict__ in,
                              unsigned short* __restrict__ out, int n)
{
  const int i = (blockIdx.x * 256 + threadIdx.x) * 4;
  if (i >= n) return;
  float4 v = *(const float4*)(in + i);
  ushort4 h = make_ushort4(f2bf(v.x), f2bf(v.y), f2bf(v.z), f2bf(v.w));
  *(ushort4*)(out + i) = h;
}

// ---------------------------------------------------------------------------
extern "C" void kernel_launch(void* const* d_in, const int* in_sizes, int n_in,
                              void* d_out, int out_size, void* d_ws, size_t ws_size,
                              hipStream_t stream) {
  (void)in_sizes; (void)n_in; (void)out_size; (void)ws_size;
  const int B = 8192, OBS = 4608;
  typedef unsigned short us;

  // JAX pytree flattening: dicts in sorted-key order.
  const float* obs     = (const float*)d_in[0];
  const float* pm_b1   = (const float*)d_in[1];   // pmoe.b1   [4,1024]
  const float* pm_b2   = (const float*)d_in[2];   // pmoe.b2   [4,1024]
  const float* pm_gb   = (const float*)d_in[3];   // pmoe.gate.b [4]
  const float* pm_gw   = (const float*)d_in[4];   // pmoe.gate.w [1024,4]
  const float* pm_w1   = (const float*)d_in[5];   // [4,1024,1024]
  const float* pm_w2   = (const float*)d_in[6];   // [4,1024,1024]
  const float* po_l1_b = (const float*)d_in[7];
  const float* po_l1_w = (const float*)d_in[8];   // [1024,512]
  const float* po_l2_b = (const float*)d_in[9];
  const float* po_l2_w = (const float*)d_in[10];  // [512,64]
  const float* r_l1_b  = (const float*)d_in[11];
  const float* r_l1_w  = (const float*)d_in[12];  // [4608,2048]
  const float* r_l2_b  = (const float*)d_in[13];
  const float* r_l2_w  = (const float*)d_in[14];  // [2048,1024]
  const float* r_l3_b  = (const float*)d_in[15];
  const float* r_l3_w  = (const float*)d_in[16];  // [1024,1024]
  const float* r_ln1_b = (const float*)d_in[17];
  const float* r_ln1_g = (const float*)d_in[18];
  const float* r_ln2_b = (const float*)d_in[19];
  const float* r_ln2_g = (const float*)d_in[20];
  const float* r_ln3_b = (const float*)d_in[21];
  const float* r_ln3_g = (const float*)d_in[22];
  const float* mo_b1   = (const float*)d_in[23];  // [4,1536]
  const float* mo_b2   = (const float*)d_in[24];  // [4,1024]
  const float* mo_gb   = (const float*)d_in[25];
  const float* mo_gw   = (const float*)d_in[26];  // [1024,4]
  const float* mo_w1   = (const float*)d_in[27];  // [4,1024,1536]
  const float* mo_w2   = (const float*)d_in[28];  // [4,1536,1024]
  const float* rb1_f1_b = (const float*)d_in[29];
  const float* rb1_f1_w = (const float*)d_in[30];
  const float* rb1_f2_b = (const float*)d_in[31];
  const float* rb1_f2_w = (const float*)d_in[32];
  const float* rb1_l1_b = (const float*)d_in[33];
  const float* rb1_l1_g = (const float*)d_in[34];
  const float* rb1_l2_b = (const float*)d_in[35];
  const float* rb1_l2_g = (const float*)d_in[36];
  const float* rb2_f1_b = (const float*)d_in[37];
  const float* rb2_f1_w = (const float*)d_in[38];
  const float* rb2_f2_b = (const float*)d_in[39];
  const float* rb2_f2_w = (const float*)d_in[40];
  const float* rb2_l1_b = (const float*)d_in[41];
  const float* rb2_l1_g = (const float*)d_in[42];
  const float* rb2_l2_b = (const float*)d_in[43];
  const float* rb2_l2_g = (const float*)d_in[44];
  const float* v_l1_b  = (const float*)d_in[45];
  const float* v_l1_w  = (const float*)d_in[46];  // [1024,256]
  const float* v_l2_b  = (const float*)d_in[47];  // [1]
  const float* v_l2_w  = (const float*)d_in[48];  // [256,1]

  // workspace carve (256B aligned)
  char* wsBase = (char*)d_ws;
  size_t cur = 0;
  auto alloc = [&](size_t bytes) -> void* {
    void* p = wsBase + cur;
    cur += (bytes + 255) & ~(size_t)255;
    return p;
  };
  us* Wt_l1    = (us*)alloc((size_t)2048 * 4608 * 2);
  us* Wt_l2    = (us*)alloc((size_t)1024 * 2048 * 2);
  us* Wt_l3    = (us*)alloc((size_t)1024 * 1024 * 2);
  us* Wt_rb1f1 = (us*)alloc((size_t)1024 * 1024 * 2);
  us* Wt_rb1f2 = (us*)alloc((size_t)1024 * 1024 * 2);
  us* Wt_rb2f1 = (us*)alloc((size_t)1024 * 1024 * 2);
  us* Wt_rb2f2 = (us*)alloc((size_t)1024 * 1024 * 2);
  us* Wt_mo1   = (us*)alloc((size_t)4 * 1536 * 1024 * 2);
  us* Wt_mo2   = (us*)alloc((size_t)4 * 1024 * 1536 * 2);
  us* Wt_pm1   = (us*)alloc((size_t)4 * 1024 * 1024 * 2);
  us* Wt_pm2   = (us*)alloc((size_t)4 * 1024 * 1024 * 2);
  us* Wt_po1   = (us*)alloc((size_t)512 * 1024 * 2);
  us* Wt_po2   = (us*)alloc((size_t)64 * 512 * 2);
  us* Wt_v1    = (us*)alloc((size_t)256 * 1024 * 2);
  us* obs_b    = (us*)alloc((size_t)B * OBS * 2);
  float* FA    = (float*)alloc((size_t)B * 2048 * 4);  // f32 gemm scratch
  float* FX    = (float*)alloc((size_t)B * 1024 * 4);  // f32 residual "x"
  us* BA       = (us*)alloc((size_t)B * 2048 * 2);     // bf16 act (wide)
  us* BB       = (us*)alloc((size_t)B * 1024 * 2);     // bf16 act
  us* BC       = (us*)alloc((size_t)B * 1024 * 2);     // bf16 act
  us* BH       = (us*)alloc((size_t)B * 1536 * 2);     // bf16 moe hidden
  int* expertId = (int*)alloc((size_t)B * 4);
  int* rowIdx   = (int*)alloc((size_t)B * 4);
  int* route    = (int*)alloc(16 * 4);   // [0..3] cnt, [4..7] off, [8..11] cursor

  float* outPolicy = (float*)d_out;
  float* outValue  = outPolicy + (size_t)B * 64;
  float* outH      = outValue + B;

  // ---- weight prep (bf16 transposed [N,K]) ----
  dim3 tpb(32, 8);
  auto T = [&](const float* in, us* out, int K, int N) {
    dim3 g((N + 31) / 32, (K + 31) / 32);
    k_transpose_w<<<g, tpb, 0, stream>>>(in, out, K, N);
  };
  T(r_l1_w, Wt_l1, 4608, 2048);
  T(r_l2_w, Wt_l2, 2048, 1024);
  T(r_l3_w, Wt_l3, 1024, 1024);
  T(rb1_f1_w, Wt_rb1f1, 1024, 1024);
  T(rb1_f2_w, Wt_rb1f2, 1024, 1024);
  T(rb2_f1_w, Wt_rb2f1, 1024, 1024);
  T(rb2_f2_w, Wt_rb2f2, 1024, 1024);
  for (int e = 0; e < 4; ++e) {
    T(mo_w1 + (size_t)e * 1024 * 1536, Wt_mo1 + (size_t)e * 1536 * 1024, 1024, 1536);
    T(mo_w2 + (size_t)e * 1536 * 1024, Wt_mo2 + (size_t)e * 1024 * 1536, 1536, 1024);
    T(pm_w1 + (size_t)e * 1024 * 1024, Wt_pm1 + (size_t)e * 1024 * 1024, 1024, 1024);
    T(pm_w2 + (size_t)e * 1024 * 1024, Wt_pm2 + (size_t)e * 1024 * 1024, 1024, 1024);
  }
  T(po_l1_w, Wt_po1, 1024, 512);
  T(po_l2_w, Wt_po2, 512, 64);
  T(v_l1_w, Wt_v1, 1024, 256);
  k_f32_to_bf16<<<(B * OBS / 4) / 256, 256, 0, stream>>>(obs, obs_b, B * OBS);

  // ---- helpers ----
  auto gemm = [&](const us* A, const us* W, const float* bias, float* oF, us* oB,
                  int M, int N, int K, bool relu) {
    dim3 g((N + 127) / 128, (M + 127) / 128, 1);
    if (relu)
      k_gemm<0, true><<<g, 256, 0, stream>>>(A, W, bias, oF, oB, M, N, K, nullptr, nullptr);
    else
      k_gemm<0, false><<<g, 256, 0, stream>>>(A, W, bias, oF, oB, M, N, K, nullptr, nullptr);
  };
  auto gemmMoeGather = [&](const us* A, const us* W, const float* bias, us* oB,
                           int N, int K) {  // relu, bf16 compact out
    dim3 g((N + 127) / 128, B / 128, 4);
    k_gemm<1, true><<<g, 256, 0, stream>>>(A, W, bias, nullptr, oB, B, N, K, rowIdx, route);
  };
  auto gemmMoeCompact = [&](const us* A, const us* W, const float* bias, float* oF,
                            int N, int K) {  // no relu, f32 compact out
    dim3 g((N + 127) / 128, B / 128, 4);
    k_gemm<2, false><<<g, 256, 0, stream>>>(A, W, bias, oF, nullptr, B, N, K, rowIdx, route);
  };
  auto ln = [&](const float* in, const float* g_, const float* b_, int D,
                float* oF, us* oB, bool relu) {
    if (relu) k_layernorm<true><<<B, 256, 0, stream>>>(in, g_, b_, D, oF, oB);
    else      k_layernorm<false><<<B, 256, 0, stream>>>(in, g_, b_, D, oF, oB);
  };
  auto routing = [&](const float* x, const float* gw, const float* gb) {
    k_route_init<<<1, 32, 0, stream>>>(route);
    k_gate<<<B / 8, 256, 0, stream>>>(x, gw, gb, 1024, expertId, route);
    k_route_scan<<<1, 32, 0, stream>>>(route);
    k_route_scatter<<<B / 256, 256, 0, stream>>>(expertId, route, rowIdx);
  };

  // ---- representation trunk ----
  gemm(obs_b, Wt_l1, r_l1_b, FA, nullptr, B, 2048, 4608, true);      // relu(l1)
  ln(FA, r_ln1_g, r_ln1_b, 2048, nullptr, BA, false);                // ln1 -> bf16
  gemm(BA, Wt_l2, r_l2_b, FA, nullptr, B, 1024, 2048, true);
  ln(FA, r_ln2_g, r_ln2_b, 1024, nullptr, BB, false);
  gemm(BB, Wt_l3, r_l3_b, FA, nullptr, B, 1024, 1024, true);
  ln(FA, r_ln3_g, r_ln3_b, 1024, FX, BB, false);                     // x3 f32+bf16

  // resblock 1
  gemm(BB, Wt_rb1f1, rb1_f1_b, FA, nullptr, B, 1024, 1024, false);
  ln(FA, rb1_l1_g, rb1_l1_b, 1024, nullptr, BB, true);               // relu(ln1)
  gemm(BB, Wt_rb1f2, rb1_f2_b, FA, nullptr, B, 1024, 1024, false);
  ln(FA, rb1_l2_g, rb1_l2_b, 1024, FA, nullptr, false);              // ln2 in place
  k_add_relu<<<B * 1024 / 1024, 256, 0, stream>>>(FA, FX, nullptr, BB, B * 1024);

  // rep MoE (top-1 routed)
  routing(FX, mo_gw, mo_gb);
  gemmMoeGather(BB, Wt_mo1, mo_b1, BH, 1536, 1024);                  // h = relu(x@w1+b1)
  gemmMoeCompact(BH, Wt_mo2, mo_b2, FA, 1024, 1536);                 // y = h@w2+b2
  k_scatter_add<<<B, 256, 0, stream>>>(FA, FX, BB, rowIdx, 1024);    // x = y + x

  // resblock 2
  gemm(BB, Wt_rb2f1, rb2_f1_b, FA, nullptr, B, 1024, 1024, false);
  ln(FA, rb2_l1_g, rb2_l1_b, 1024, nullptr, BB, true);
  gemm(BB, Wt_rb2f2, rb2_f2_b, FA, nullptr, B, 1024, 1024, false);
  ln(FA, rb2_l2_g, rb2_l2_b, 1024, FA, nullptr, false);
  k_add_relu<<<B * 1024 / 1024, 256, 0, stream>>>(FA, FX, outH, BB, B * 1024); // h -> d_out

  // policy MoE
  routing(FX, pm_gw, pm_gb);
  gemmMoeGather(BB, Wt_pm1, pm_b1, BH, 1024, 1024);
  gemmMoeCompact(BH, Wt_pm2, pm_b2, FA, 1024, 1024);
  k_scatter_add<<<B, 256, 0, stream>>>(FA, FX, BC, rowIdx, 1024);    // pl -> BC bf16

  // heads
  gemm(BC, Wt_po1, po_l1_b, nullptr, BH, B, 512, 1024, true);        // relu(l1) bf16
  gemm(BH, Wt_po2, po_l2_b, outPolicy, nullptr, B, 64, 512, false);  // policy logits
  gemm(BB, Wt_v1, v_l1_b, nullptr, BC, B, 256, 1024, true);          // value hidden
  k_rowdot<<<B / 8, 256, 0, stream>>>(BC, v_l2_w, v_l2_b, outValue, 256);
}
